// MemoryLayer_20220706030247
// MI455X (gfx1250) — matrix operand order repaired and checked by reading, not compiled, not run on hardware
//
#include <hip/hip_runtime.h>

// ---------------------------------------------------------------------------
// MI455X (gfx1250, wave32) implementation of the product-key memory layer.
//   K0: one-shot W1/W2 fp32 -> bf16 pre-convert (removes cvt from hot loop)
//   K1: fused  h = relu(x@W1^T+b1) ; q = h@W2^T+b2   (bf16 WMMA, f32 accum,
//       software-pipelined B-fragment double buffering)
//   K2: per-(token,head) scores + two-stage top-8 + softmax (wave32 shuffles)
//   K3: weighted gather of values rows (pure HBM bandwidth, b128 loads)
// ---------------------------------------------------------------------------

typedef __attribute__((ext_vector_type(16))) __bf16 v16bf;
typedef __attribute__((ext_vector_type(8)))  float  v8f;

#define T_TOKENS 8192
#define DDIM     1024
#define H1DIM    128
#define QDIM     32      // HEADS * K_DIM
#define NKEYS    256
#define KNN      8
#define NEG_INF  (-3.0e38f)

// -------- A-side fragment loader: fp32 memory -> bf16 fragment (on the fly)
__device__ __forceinline__ v16bf cvt16(const float* __restrict__ p0,
                                       const float* __restrict__ p1) {
  float4 a = *(const float4*)(p0);
  float4 b = *(const float4*)(p0 + 4);
  float4 c = *(const float4*)(p1);
  float4 d = *(const float4*)(p1 + 4);
  v16bf r;
  r[0]  = (__bf16)a.x; r[1]  = (__bf16)a.y; r[2]  = (__bf16)a.z; r[3]  = (__bf16)a.w;
  r[4]  = (__bf16)b.x; r[5]  = (__bf16)b.y; r[6]  = (__bf16)b.z; r[7]  = (__bf16)b.w;
  r[8]  = (__bf16)c.x; r[9]  = (__bf16)c.y; r[10] = (__bf16)c.z; r[11] = (__bf16)c.w;
  r[12] = (__bf16)d.x; r[13] = (__bf16)d.y; r[14] = (__bf16)d.z; r[15] = (__bf16)d.w;
  return r;
}

// -------- B-side fragment loader: 16 contiguous bf16 (32B), two b128 loads
__device__ __forceinline__ v16bf load16bf(const __bf16* __restrict__ p) {
  uint4 a = ((const uint4*)p)[0];
  uint4 b = ((const uint4*)p)[1];
  v16bf r;
  __builtin_memcpy((char*)&r,      &a, 16);
  __builtin_memcpy((char*)&r + 16, &b, 16);
  return r;
}

// ---------------------------------------------------------------------------
// Kernel 0: pre-convert W1 (128x1024) and W2 (32x128) to bf16 in workspace.
// ---------------------------------------------------------------------------
__global__ __launch_bounds__(256) void cvt_weights_kernel(
    const float* __restrict__ W1, const float* __restrict__ W2,
    __bf16* __restrict__ w1bf, __bf16* __restrict__ w2bf) {
  const int i = blockIdx.x * 256 + threadIdx.x;
  if (i < H1DIM * DDIM) w1bf[i] = (__bf16)W1[i];
  if (i < QDIM * H1DIM) w2bf[i] = (__bf16)W2[i];
}

// ---------------------------------------------------------------------------
// Kernel 1: fused two-layer MLP producing q (T x 32, fp32) via bf16 WMMA.
// 4 waves/block, each wave owns a 16-token strip. GEMM1 keeps the full
// 16x128 h-tile in accumulators; h is relayouted C->A through LDS (bf16).
// ---------------------------------------------------------------------------
#define LDS_HSTRIDE 136   // 128 + 8 bf16 pad; row stride 272B, 16B aligned

__global__ __launch_bounds__(128) void fused_mlp_kernel(
    const float* __restrict__ x, const __bf16* __restrict__ w1bf,
    const float* __restrict__ b1, const __bf16* __restrict__ w2bf,
    const float* __restrict__ b2, float* __restrict__ q_out) {
  __shared__ __bf16 lds_h[4][16 * LDS_HSTRIDE];

  const int lane  = threadIdx.x & 31;
  const int wave  = threadIdx.x >> 5;
  const int tbase = (blockIdx.x * 4 + wave) * 16;
  const int mrow  = lane & 15;       // M (A row) or N column for B/C frags
  const int khalf = lane >> 4;       // which K-half this lane carries

  // ---------------- GEMM1: h = relu(x @ W1^T + b1), 16 x 128 tile ----------
  v8f acc[8];
#pragma unroll
  for (int i = 0; i < 8; ++i) acc[i] = (v8f){};

  const float* xrow = x + (size_t)(tbase + mrow) * DDIM;
#pragma unroll 4
  for (int ks = 0; ks < DDIM / 32; ++ks) {
    const int kb = ks * 32;
    if (ks + 1 < DDIM / 32)  // global_prefetch_b8, WGP scope (all levels)
      __builtin_prefetch(xrow + kb + 32 + khalf * 8, 0, 3);

    // A fragment (16x32 bf16): lane half 0 -> K 0..15, half 1 -> K 16..31
    const float* ap = xrow + kb + khalf * 16;
    v16bf afrag = cvt16(ap, ap + 8);

    // B fragments (32x16 bf16): B[k][n] = W1[n][k]; lane halves K 0..15 / 16..31.
    // Double-buffered so each WMMA's wait covers a load issued one tile ago.
    const __bf16* wbase = w1bf + (size_t)mrow * DDIM + kb + khalf * 16;
    v16bf bfrag = load16bf(wbase);
#pragma unroll
    for (int nt = 0; nt < 8; ++nt) {
      v16bf bnext = (nt < 7) ? load16bf(wbase + (size_t)(nt + 1) * 16 * DDIM)
                             : bfrag;
      acc[nt] = __builtin_amdgcn_wmma_f32_16x16x32_bf16(
          false, afrag, false, bfrag, (short)0, acc[nt], false, false);
      bfrag = bnext;
    }
  }

  // bias + relu, spill 16x128 h tile to LDS as bf16 (C-layout -> memory)
  __bf16* hbase = &lds_h[wave][0];
#pragma unroll
  for (int nt = 0; nt < 8; ++nt) {
    const float bn = b1[nt * 16 + mrow];
#pragma unroll
    for (int r = 0; r < 8; ++r) {
      float v = fmaxf(acc[nt][r] + bn, 0.0f);
      const int M = r + khalf * 8;
      hbase[M * LDS_HSTRIDE + nt * 16 + mrow] = (__bf16)v;
    }
  }
  __syncthreads();

  // ---------------- GEMM2: q = h @ W2^T + b2, 16 x 32 ----------------------
  v8f qacc[2];
  qacc[0] = (v8f){};
  qacc[1] = (v8f){};
#pragma unroll
  for (int k2 = 0; k2 < 4; ++k2) {
    // A fragment of h from LDS, contiguous K halves (16B-aligned ds_load_b128)
    const __bf16* hp = hbase + mrow * LDS_HSTRIDE + k2 * 32 + khalf * 16;
    uint4 lo = *(const uint4*)(hp);
    uint4 hi = *(const uint4*)(hp + 8);
    v16bf a2;
    __builtin_memcpy((char*)&a2,      &lo, 16);
    __builtin_memcpy((char*)&a2 + 16, &hi, 16);
#pragma unroll
    for (int n2 = 0; n2 < 2; ++n2) {
      v16bf b2f = load16bf(w2bf + (size_t)(n2 * 16 + mrow) * H1DIM + k2 * 32 + khalf * 16);
      qacc[n2] = __builtin_amdgcn_wmma_f32_16x16x32_bf16(
          false, a2, false, b2f, (short)0, qacc[n2], false, false);
    }
  }

#pragma unroll
  for (int n2 = 0; n2 < 2; ++n2) {
    const int n   = n2 * 16 + mrow;
    const float bb = b2[n];
#pragma unroll
    for (int r = 0; r < 8; ++r) {
      const int M = r + khalf * 8;
      q_out[(size_t)(tbase + M) * QDIM + n] = qacc[n2][r] + bb;
    }
  }
}

// ---------------------------------------------------------------------------
// Kernel 2: scores + two-stage top-8 + softmax. One wave32 per (token, head).
// 256 keys -> 8 per lane; argmax via 5-step butterfly shuffle, 8 iterations.
// ---------------------------------------------------------------------------
__device__ __forceinline__ void wave_top8(float s[8], int lane,
                                          float sc[8], int idx[8]) {
  for (int it = 0; it < KNN; ++it) {
    float lm = NEG_INF;
    int   ln = 0x7fffffff;
#pragma unroll
    for (int j = 0; j < 8; ++j) {
      if (s[j] > lm) { lm = s[j]; ln = j * 32 + lane; }
    }
#pragma unroll
    for (int off = 16; off > 0; off >>= 1) {
      float ov = __shfl_xor(lm, off, 32);
      int   on = __shfl_xor(ln, off, 32);
      if (ov > lm || (ov == lm && on < ln)) { lm = ov; ln = on; }
    }
    sc[it]  = lm;
    idx[it] = ln;
    if ((ln & 31) == lane) s[ln >> 5] = NEG_INF;  // retire winner
  }
}

__global__ __launch_bounds__(128) void topk_kernel(
    const float* __restrict__ q_ws, const float* __restrict__ keys,
    int* __restrict__ idx_ws, float* __restrict__ w_ws) {
  const int gtid = blockIdx.x * blockDim.x + threadIdx.x;
  const int wid  = gtid >> 5;        // 0 .. T*HEADS-1
  const int lane = threadIdx.x & 31;
  const int t    = wid >> 1;
  const int head = wid & 1;

  const float* qp = q_ws + (size_t)t * QDIM + head * 16;
  float q1[8], q2[8];
#pragma unroll
  for (int c = 0; c < 8; ++c) { q1[c] = qp[c]; q2[c] = qp[8 + c]; }

  // sub-key scores: 8 keys per lane, strided n = j*32+lane
  float s1[8], s2[8];
#pragma unroll
  for (int j = 0; j < 8; ++j) {
    const int n = j * 32 + lane;
    const float* k1 = keys + (((size_t)head * 2 + 0) * NKEYS + n) * 8;
    const float* k2 = keys + (((size_t)head * 2 + 1) * NKEYS + n) * 8;
    float a = 0.f, b = 0.f;
#pragma unroll
    for (int c = 0; c < 8; ++c) { a += q1[c] * k1[c]; b += q2[c] * k2[c]; }
    s1[j] = a; s2[j] = b;
  }

  float sc1[8], sc2[8]; int i1[8], i2[8];
  wave_top8(s1, lane, sc1, i1);
  wave_top8(s2, lane, sc2, i2);

  // 64 combos, 2 per lane, second-stage top-8
  float cs0 = sc1[lane >> 3]       + sc2[lane & 7];
  float cs1 = sc1[(lane >> 3) + 4] + sc2[lane & 7];
  float scf[8]; int cidf[8];
  for (int it = 0; it < KNN; ++it) {
    float lm = cs0; int lc = lane;
    if (cs1 > lm) { lm = cs1; lc = lane + 32; }
#pragma unroll
    for (int off = 16; off > 0; off >>= 1) {
      float ov = __shfl_xor(lm, off, 32);
      int   oc = __shfl_xor(lc, off, 32);
      if (ov > lm || (ov == lm && oc < lc)) { lm = ov; lc = oc; }
    }
    scf[it] = lm; cidf[it] = lc;
    if (lc == lane)           cs0 = NEG_INF;
    else if (lc == lane + 32) cs1 = NEG_INF;
  }

  // softmax over the 8 winners (scf[0] is the max by construction)
  float e[8], sum = 0.f;
#pragma unroll
  for (int k = 0; k < KNN; ++k) { e[k] = __expf(scf[k] - scf[0]); sum += e[k]; }
  const float inv = 1.0f / sum;

  if (lane < KNN) {
    const int c   = cidf[lane];
    const int idx = i1[c >> 3] * NKEYS + i2[c & 7];
    const size_t o = ((size_t)t * 2 + head) * KNN + lane;
    idx_ws[o] = idx;
    w_ws[o]   = e[lane] * inv;
  }
}

// ---------------------------------------------------------------------------
// Kernel 3: out[t] = sum_{j<16} w[j] * values[idx[j]]  — HBM-bound gather.
// One token per 256-thread block; each row read is a coalesced 4KB b128 stream.
// ---------------------------------------------------------------------------
__global__ __launch_bounds__(256) void gather_kernel(
    const float* __restrict__ values, const int* __restrict__ idx_ws,
    const float* __restrict__ w_ws, float* __restrict__ out) {
  __shared__ int   sIdx[16];
  __shared__ float sW[16];
  const int t = blockIdx.x;
  if (threadIdx.x < 16) {
    sIdx[threadIdx.x] = idx_ws[(size_t)t * 16 + threadIdx.x];
    sW[threadIdx.x]   = w_ws[(size_t)t * 16 + threadIdx.x];
  }
  __syncthreads();

  const int c = threadIdx.x;  // 4 floats per thread
  float4 acc = make_float4(0.f, 0.f, 0.f, 0.f);
#pragma unroll 4
  for (int j = 0; j < 16; ++j) {
    const float4* vp = (const float4*)(values + (size_t)sIdx[j] * DDIM) + c;
    const float4 v = *vp;
    const float  w = sW[j];
    acc.x += w * v.x; acc.y += w * v.y; acc.z += w * v.z; acc.w += w * v.w;
  }
  ((float4*)(out + (size_t)t * DDIM))[c] = acc;
}

// ---------------------------------------------------------------------------
extern "C" void kernel_launch(void* const* d_in, const int* in_sizes, int n_in,
                              void* d_out, int out_size, void* d_ws, size_t ws_size,
                              hipStream_t stream) {
  const float* x      = (const float*)d_in[0];
  const float* W1     = (const float*)d_in[1];
  const float* b1     = (const float*)d_in[2];
  const float* W2     = (const float*)d_in[3];
  const float* b2     = (const float*)d_in[4];
  const float* keys   = (const float*)d_in[5];
  const float* values = (const float*)d_in[6];
  float* out = (float*)d_out;

  char*   ws     = (char*)d_ws;
  float*  q_ws   = (float*)ws;                                     // 1 MiB
  int*    idx_ws = (int*)(ws + (size_t)(1 << 20));                 // 512 KiB
  float*  w_ws   = (float*)(ws + (size_t)(1 << 20) + (512 << 10)); // 512 KiB
  __bf16* w1bf   = (__bf16*)(ws + (size_t)(2 << 20));              // 256 KiB
  __bf16* w2bf   = (__bf16*)(ws + (size_t)(2 << 20) + (256 << 10));// 8 KiB

  // K0: weight pre-convert (131072 elements -> 512 blocks of 256)
  cvt_weights_kernel<<<512, 256, 0, stream>>>(W1, W2, w1bf, w2bf);
  // K1: 8192 tokens / (4 waves * 16 tokens) = 128 blocks of 128 threads
  fused_mlp_kernel<<<128, 128, 0, stream>>>(x, w1bf, b1, w2bf, b2, q_ws);
  // K2: 8192*2 (token,head) waves / 4 waves per block = 4096 blocks
  topk_kernel<<<4096, 128, 0, stream>>>(q_ws, keys, idx_ws, w_ws);
  // K3: one token per block
  gather_kernel<<<T_TOKENS, 256, 0, stream>>>(values, idx_ws, w_ws, out);
}
